// MixedScoreMultiHeadAttention_70068096467030
// MI455X (gfx1250) — compile-verified, matrix operand order checked by reading, not run on hardware
//
#include <hip/hip_runtime.h>

// ---------------------------------------------------------------------------
// MixedScoreMultiHeadAttention on MI455X (gfx1250), wave32.
// bf16-resident pipeline:
//   [cvt] fp32->bf16 (embeddings; weights transposed to [N][K])
//   [gemm_bb x3] Q/K/V projections (V written transposed [b][d][c])
//   [attn] fused QK^T + score-MLP + softmax + probs*V (all WMMA)
//   [gemm_bb] out @ Wo -> f32
// All matmuls: v_wmma_f32_16x16x32_bf16.  All LDS staging: async-to-LDS b128
// with double buffering (ASYNCcnt / s_wait_asynccnt).
// ---------------------------------------------------------------------------

#define B_TOT 8
#define R_TOT 256
#define C_TOT 256
#define HEADS 8
#define DHEAD 64
#define EMB   512

typedef __attribute__((ext_vector_type(16))) __bf16 v16bf;
typedef __attribute__((ext_vector_type(8)))  float  v8f;
typedef __attribute__((ext_vector_type(4)))  unsigned int v4u;

#if defined(__has_builtin)
#  if __has_builtin(__builtin_amdgcn_global_load_async_to_lds_b128)
#    define HAVE_ASYNC_BUILTIN 1
#  endif
#endif

// Builtin signature (from hipcc diagnostic): param0 = global (addrspace(1))
// int4*, param1 = LDS (addrspace(3)) int4*, then imm offset, imm cpol.
typedef __attribute__((vector_size(16))) int i32x4;
typedef __attribute__((address_space(1))) i32x4* gp128_t;
typedef __attribute__((address_space(3))) i32x4* lp128_t;

// 16-byte global -> LDS async copy (per active lane), tracked by ASYNCcnt.
static __device__ __forceinline__ void cp16_async(void* lds, const void* g) {
#ifdef HAVE_ASYNC_BUILTIN
  __builtin_amdgcn_global_load_async_to_lds_b128(
      (gp128_t)(unsigned long long)(size_t)g,
      (lp128_t)(unsigned)(size_t)lds, 0, 0);
#else
  unsigned l = (unsigned)(size_t)lds;
  unsigned long long ga = (unsigned long long)(size_t)g;
  asm volatile("global_load_async_to_lds_b128 %0, %1, off"
               :: "v"(l), "v"(ga) : "memory");
#endif
}
static __device__ __forceinline__ void async_wait0() {
  asm volatile("s_wait_asynccnt 0x0" ::: "memory");
}

// Hardware bf16 conversion (v_cvt_*), not bit-twiddling.
static __device__ __forceinline__ unsigned short f2bfbits(float f) {
  union { __bf16 h; unsigned short u; } x;
  x.h = (__bf16)f;
  return x.u;
}
static __device__ __forceinline__ float bf2f(unsigned short u) {
  union { unsigned short u; __bf16 h; } x;
  x.u = u;
  return (float)x.h;
}
static __device__ __forceinline__ v8f zero8() {
  v8f z = {0.f, 0.f, 0.f, 0.f, 0.f, 0.f, 0.f, 0.f};
  return z;
}
static __device__ __forceinline__ v8f wmma_bf16(v16bf a, v16bf b, v8f c) {
  return __builtin_amdgcn_wmma_f32_16x16x32_bf16(false, a, false, b,
                                                 (short)0, c, false, false);
}

// 16x32 bf16 fragment from LDS tile stored row-major [row][k].
// wave32 16-bit layout: lane&15 = row; lane<16 -> K {0..7,16..23},
// lane>=16 -> K {8..15,24..31}.  Two ds_read_b128 per fragment.
static __device__ __forceinline__ v16bf load_frag(const unsigned short* base,
                                                  int ld, int row0, int k0,
                                                  int lane) {
  int r  = row0 + (lane & 15);
  int kb = k0 + ((lane < 16) ? 0 : 8);
  const unsigned short* p = base + r * ld + kb;
  union { v4u q[2]; v16bf v; } u;
  u.q[0] = *(const v4u*)(p);
  u.q[1] = *(const v4u*)(p + 16);
  return u.v;
}

// ---------------------------------------------------------------------------
// Elementwise fp32 -> bf16
__global__ __launch_bounds__(256) void cvt_bf16_kernel(
    const float* __restrict__ src, unsigned short* __restrict__ dst, int n) {
  int i = blockIdx.x * 256 + threadIdx.x;
  if (i < n) dst[i] = f2bfbits(src[i]);
}
// fp32 [K][N] -> bf16 [N][K] (weights pre-transposed for [n][k] B-fragments)
__global__ __launch_bounds__(256) void transpose_cvt_kernel(
    const float* __restrict__ W, unsigned short* __restrict__ Wt, int K, int N) {
  int i = blockIdx.x * 256 + threadIdx.x;
  if (i < K * N) {
    int k = i / N, n = i % N;
    Wt[(size_t)n * K + k] = f2bfbits(W[i]);
  }
}

// ---------------------------------------------------------------------------
// GEMM: out[M,N] = A[M,K] @ Bt[N,K]^T, bf16 in, f32 accum.
// Block tile 128x128x32, 8 waves each owning 64x32, async double-buffered LDS.
// OUT_MODE: 0 = f32 row-major, 1 = bf16 row-major,
//           2 = bf16 V^T: row m=(b,c), col n=d -> out[(b*EMB+n)*C_TOT+c]
// ---------------------------------------------------------------------------
template <int OUT_MODE>
__global__ __launch_bounds__(256) void gemm_bb(
    const unsigned short* __restrict__ A, const unsigned short* __restrict__ Bt,
    void* __restrict__ out, int M, int N, int K) {
  __shared__ unsigned short sA[2][128 * 40];   // [m][k], stride 40
  __shared__ unsigned short sB[2][128 * 40];   // [n][k], stride 40

  const int tid = threadIdx.x;
  const int lane = tid & 31;
  const int wv = tid >> 5;
  const int wm = (wv >> 2) * 64;
  const int wn = (wv & 3) * 32;
  const int m0 = blockIdx.y * 128;
  const int n0 = blockIdx.x * 128;

  auto stage = [&](int buf, int k0) {
    for (int i = tid; i < 512; i += 256) {          // 128 rows x 4 chunks
      int r = i >> 2, ck = (i & 3) * 8;
      cp16_async(&sA[buf][r * 40 + ck], A + (size_t)(m0 + r) * K + k0 + ck);
    }
    for (int i = tid; i < 512; i += 256) {
      int r = i >> 2, ck = (i & 3) * 8;
      cp16_async(&sB[buf][r * 40 + ck], Bt + (size_t)(n0 + r) * K + k0 + ck);
    }
  };

  v8f acc[4][2];
#pragma unroll
  for (int s = 0; s < 4; ++s)
#pragma unroll
    for (int t = 0; t < 2; ++t) acc[s][t] = zero8();

  stage(0, 0);
  async_wait0();
  __syncthreads();

  const int KT = K / 32;
  for (int kt = 0; kt < KT; ++kt) {
    const int cur = kt & 1;
    if (kt + 1 < KT) stage(cur ^ 1, (kt + 1) * 32);  // overlap with compute

    v16bf af[4], bfr[2];
#pragma unroll
    for (int s = 0; s < 4; ++s)
      af[s] = load_frag(sA[cur], 40, wm + s * 16, 0, lane);
#pragma unroll
    for (int t = 0; t < 2; ++t)
      bfr[t] = load_frag(sB[cur], 40, wn + t * 16, 0, lane);
#pragma unroll
    for (int s = 0; s < 4; ++s)
#pragma unroll
      for (int t = 0; t < 2; ++t)
        acc[s][t] = wmma_bf16(af[s], bfr[t], acc[s][t]);

    async_wait0();
    __syncthreads();
  }

  const int rr = (lane < 16) ? 0 : 8;
  const int nn = lane & 15;
#pragma unroll
  for (int s = 0; s < 4; ++s)
#pragma unroll
    for (int t = 0; t < 2; ++t)
#pragma unroll
      for (int i = 0; i < 8; ++i) {
        int m = m0 + wm + s * 16 + i + rr;
        int n = n0 + wn + t * 16 + nn;
        float v = acc[s][t][i];
        if (OUT_MODE == 0)
          ((float*)out)[(size_t)m * N + n] = v;
        else if (OUT_MODE == 1)
          ((unsigned short*)out)[(size_t)m * N + n] = f2bfbits(v);
        else {  // V^T
          int b = m >> 8, c = m & 255;
          ((unsigned short*)out)[((size_t)b * EMB + n) * C_TOT + c] =
              f2bfbits(v);
        }
      }
}

// ---------------------------------------------------------------------------
// Fused attention: one block per (batch, 16-row tile); 8 waves <-> 8 heads.
// ---------------------------------------------------------------------------
// LDS carve (ushort elements); every region 16B aligned.
#define SQ_ELEMS   (HEADS * 16 * 72)       // Q tile [h][r][d], stride 72
#define SK_ELEMS   (2 * HEADS * 16 * 72)   // K tile, double buffered
#define SW1_ELEMS  (128 * 40)              // W1^T pad [n=128][k], stride 40
#define SW2_ELEMS  (16 * 136)              // W2^T pad [n=16][k=128], stride 136
#define SLOG_ELEMS (HEADS * 16 * 256)      // logits/probs bf16 [h][r][c]
#define SS_ELEMS   (256 * 40)              // scores [pos][k=16 pad 32]
#define SU_ELEMS   (2 * HEADS * DHEAD * 40)  // hidden [pos][136] / 2x V^T bufs
#define ATTN_SMEM_BYTES                                                    \
  (2 * (SQ_ELEMS + SK_ELEMS + SW1_ELEMS + SW2_ELEMS + SLOG_ELEMS +         \
        SS_ELEMS + SU_ELEMS) + 16 * 256 /*mask*/ + 32 /*alpha*/)

__global__ __launch_bounds__(256) void attn_kernel(
    const unsigned short* __restrict__ qw,   // bf16 [B][R][H*D]
    const unsigned short* __restrict__ kw,   // bf16 [B][C][H*D]
    const unsigned short* __restrict__ vt,   // bf16 [B][H*D][C]  (V^T)
    const float* __restrict__ cost,          // f32  [B][R][C][1]
    const unsigned char* __restrict__ mask,  // bool [B][R][C]
    const float* __restrict__ W1,            // f32  [16][128]
    const float* __restrict__ W2,            // f32  [128][8]
    const float* __restrict__ alpha,         // f32  [8]
    unsigned short* __restrict__ attn_out) { // bf16 [B][R][H*D]
  extern __shared__ char smem[];
  unsigned short* sQ   = (unsigned short*)smem;
  unsigned short* sK   = sQ + SQ_ELEMS;      // sK + buf * 9216
  unsigned short* sW1  = sK + SK_ELEMS;
  unsigned short* sW2  = sW1 + SW1_ELEMS;
  unsigned short* sLog = sW2 + SW2_ELEMS;
  unsigned short* sS   = sLog + SLOG_ELEMS;
  unsigned short* sU   = sS + SS_ELEMS;      // hidden; pass2: sU + buf * 20480
  unsigned char*  sMask  = (unsigned char*)(sU + SU_ELEMS);
  float*          sAlpha = (float*)(sMask + 16 * 256);

  const int tid = threadIdx.x;
  const int lane = tid & 31;
  const int wv = tid >> 5;                 // wave == head
  const int b = blockIdx.y;
  const int r0 = blockIdx.x * 16;
  const float NEG_INF = -__builtin_inff();

  auto stage_k = [&](unsigned short* dst, int c0) {   // 128 rows x 8 chunks
    for (int i = tid; i < 1024; i += 256) {
      int row = i >> 3, ck = (i & 7) * 8;             // row = h*16 + c
      int h = row >> 4, c = row & 15;
      cp16_async(dst + row * 72 + ck,
                 kw + (size_t)(b * C_TOT + c0 + c) * EMB + h * DHEAD + ck);
    }
  };
  auto stage_v = [&](unsigned short* dst, int ch) {   // 512 rows x 4 chunks
    for (int i = tid; i < 2048; i += 256) {
      int row = i >> 2, ck = (i & 3) * 8;             // row = h*64 + d
      cp16_async(dst + row * 40 + ck,
                 vt + ((size_t)b * EMB + row) * C_TOT + ch * 32 + ck);
    }
  };

  // ---- one-time staging (all async) ----
  for (int i = tid; i < 1024; i += 256) {             // Q tile
    int row = i >> 3, ck = (i & 7) * 8;
    int h = row >> 4, r = row & 15;
    cp16_async(sQ + row * 72 + ck,
               qw + (size_t)(b * R_TOT + r0 + r) * EMB + h * DHEAD + ck);
  }
  {                                                   // mask: 16 rows x 256 B
    int row = tid >> 4, ck = (tid & 15) * 16;
    cp16_async(sMask + tid * 16,
               mask + (size_t)(b * R_TOT + r0 + row) * C_TOT + ck);
  }
  stage_k(sK, 0);
  // small weights via VALU converts
  for (int i = tid; i < SW1_ELEMS; i += 256) {
    int n = i / 40, k = i % 40;
    sW1[i] = (k < 16) ? f2bfbits(W1[k * 128 + n]) : (unsigned short)0;
  }
  for (int i = tid; i < SW2_ELEMS; i += 256) {
    int n = i / 136, k = i % 136;
    sW2[i] = (n < 8 && k < 128) ? f2bfbits(W2[k * 8 + n]) : (unsigned short)0;
  }
  if (tid < 8) sAlpha[tid] = alpha[tid];
  for (int k = 16; k < 40; ++k) sS[tid * 40 + k] = 0;   // K-pad stays zero
  async_wait0();
  __syncthreads();

  const float norm = 0.125f;               // 1/sqrt(64)

  // ---- pass 1: logits[h][r][c]; K tile double-buffered over the MLP ----
  for (int ct = 0; ct < 16; ++ct) {
    const int c0 = ct * 16;
    unsigned short* kcur = sK + (ct & 1) * 9216;
    if (ct + 1 < 16) stage_k(sK + ((ct & 1) ^ 1) * 9216, c0 + 16);

    // (a) per-head QK^T
    {
      v8f acc = zero8();
      const unsigned short* qh = sQ + wv * 16 * 72;
      const unsigned short* kh = kcur + wv * 16 * 72;
#pragma unroll
      for (int kk = 0; kk < DHEAD; kk += 32) {
        v16bf a = load_frag(qh, 72, 0, kk, lane);
        v16bf bb = load_frag(kh, 72, 0, kk, lane);
        acc = wmma_bf16(a, bb, acc);
      }
      int rr = (lane < 16) ? 0 : 8, cc = lane & 15;
#pragma unroll
      for (int i = 0; i < 8; ++i)
        sS[((i + rr) * 16 + cc) * 40 + 2 * wv] = f2bfbits(norm * acc[i]);
    }
    // cost features: thread <-> position
    {
      int r = tid >> 4, cc = tid & 15;
      float cv = cost[(size_t)(b * R_TOT + r0 + r) * C_TOT + c0 + cc];
#pragma unroll
      for (int h = 0; h < 8; ++h)
        sS[tid * 40 + 2 * h + 1] = f2bfbits(sAlpha[h] * cv);
    }
    __syncthreads();

    // (b) layer 1: hidden = relu(sS[256x32] @ W1p[32x128])
#pragma unroll
    for (int mi = 0; mi < 2; ++mi) {
      int mt = wv * 2 + mi;
      v16bf a = load_frag(sS, 40, mt * 16, 0, lane);
#pragma unroll
      for (int nt = 0; nt < 8; ++nt) {
        v16bf bb = load_frag(sW1, 40, nt * 16, 0, lane);
        v8f acc = wmma_bf16(a, bb, zero8());
        int rr = (lane < 16) ? 0 : 8, nn = nt * 16 + (lane & 15);
#pragma unroll
        for (int i = 0; i < 8; ++i) {
          float hv = acc[i];
          sU[(mt * 16 + i + rr) * 136 + nn] = f2bfbits(hv > 0.f ? hv : 0.f);
        }
      }
    }
    __syncthreads();

    // (c) layer 2: logits = hidden[256x128] @ W2p[128x16], masked
#pragma unroll
    for (int mi = 0; mi < 2; ++mi) {
      int mt = wv * 2 + mi;
      v8f acc = zero8();
#pragma unroll
      for (int ks = 0; ks < 4; ++ks) {
        v16bf a = load_frag(sU, 136, mt * 16, ks * 32, lane);
        v16bf bb = load_frag(sW2, 136, 0, ks * 32, lane);
        acc = wmma_bf16(a, bb, acc);
      }
      int rr = (lane < 16) ? 0 : 8, h = lane & 15;
      if (h < 8) {
#pragma unroll
        for (int i = 0; i < 8; ++i) {
          int pos = mt * 16 + i + rr;
          int r = pos >> 4, c = c0 + (pos & 15);
          float lg = acc[i];
          if (sMask[r * 256 + c]) lg = NEG_INF;     // where(mask, -inf, x)
          sLog[(h * 16 + r) * 256 + c] = f2bfbits(lg);
        }
      }
    }
    async_wait0();     // next K tile resident before it becomes 'cur'
    __syncthreads();
  }

  // ---- softmax over c per (h, r); all-masked rows -> 0 ----
  if (tid < 128) {
    int h = tid >> 4, r = tid & 15;
    unsigned short* row = sLog + (h * 16 + r) * 256;
    float m = NEG_INF;
    for (int c = 0; c < 256; ++c) { float x = bf2f(row[c]); m = x > m ? x : m; }
    if (m == NEG_INF) {
      for (int c = 0; c < 256; ++c) row[c] = 0;
    } else {
      float s = 0.f;
      for (int c = 0; c < 256; ++c) s += __expf(bf2f(row[c]) - m);
      float inv = 1.f / s;
      for (int c = 0; c < 256; ++c)
        row[c] = f2bfbits(__expf(bf2f(row[c]) - m) * inv);
    }
  }
  __syncthreads();

  // ---- pass 2: out[h][r][d] = probs @ V, double-buffered V^T chunks ----
  v8f oacc[4] = {zero8(), zero8(), zero8(), zero8()};
  stage_v(sU, 0);
  async_wait0();
  __syncthreads();
  for (int ch = 0; ch < 8; ++ch) {
    unsigned short* vcur = sU + (ch & 1) * 20480;
    if (ch + 1 < 8) stage_v(sU + ((ch & 1) ^ 1) * 20480, ch + 1);
    const unsigned short* prow = sLog + wv * 16 * 256;
    const unsigned short* vh = vcur + wv * 64 * 40;
    v16bf a = load_frag(prow, 256, 0, ch * 32, lane);
#pragma unroll
    for (int nt = 0; nt < 4; ++nt) {
      v16bf bb = load_frag(vh, 40, nt * 16, 0, lane);
      oacc[nt] = wmma_bf16(a, bb, oacc[nt]);
    }
    async_wait0();
    __syncthreads();
  }
  {
    int rr = (lane < 16) ? 0 : 8, nn = lane & 15;
#pragma unroll
    for (int nt = 0; nt < 4; ++nt)
#pragma unroll
      for (int i = 0; i < 8; ++i) {
        int r = i + rr, d = nt * 16 + nn;
        attn_out[(size_t)(b * R_TOT + r0 + r) * EMB + wv * DHEAD + d] =
            f2bfbits(oacc[nt][i]);
      }
  }
}

// ---------------------------------------------------------------------------
extern "C" void kernel_launch(void* const* d_in, const int* in_sizes, int n_in,
                              void* d_out, int out_size, void* d_ws,
                              size_t ws_size, hipStream_t stream) {
  const float* row_emb = (const float*)d_in[0];
  const float* col_emb = (const float*)d_in[1];
  const float* cost    = (const float*)d_in[2];
  const unsigned char* mask = (const unsigned char*)d_in[3];
  const float* Wq = (const float*)d_in[4];
  const float* Wk = (const float*)d_in[5];
  const float* Wv = (const float*)d_in[6];
  const float* Wo = (const float*)d_in[7];
  const float* W1 = (const float*)d_in[8];
  const float* W2 = (const float*)d_in[9];
  const float* alpha = (const float*)d_in[10];

  const int M = B_TOT * R_TOT;                 // 2048
  const size_t NE = (size_t)M * EMB;           // 1M elems
  const size_t NW = (size_t)EMB * EMB;         // 256K elems

  unsigned short* rowe = (unsigned short*)d_ws;   // bf16 [B][R][512]
  unsigned short* cole = rowe + NE;               // bf16 [B][C][512]
  unsigned short* WqT  = cole + NE;               // bf16 [512][512] (W^T)
  unsigned short* WkT  = WqT + NW;
  unsigned short* WvT  = WkT + NW;
  unsigned short* WoT  = WvT + NW;
  unsigned short* qw   = WoT + NW;                // bf16 [B][R][512]
  unsigned short* kwb  = qw + NE;                 // bf16 [B][C][512]
  unsigned short* vtw  = kwb + NE;                // bf16 [B][512][256] (V^T)
  unsigned short* attb = vtw + NE;                // bf16 [B][R][512]

  cvt_bf16_kernel<<<(int)(NE / 256), 256, 0, stream>>>(row_emb, rowe, (int)NE);
  cvt_bf16_kernel<<<(int)(NE / 256), 256, 0, stream>>>(col_emb, cole, (int)NE);
  transpose_cvt_kernel<<<(int)(NW / 256), 256, 0, stream>>>(Wq, WqT, EMB, EMB);
  transpose_cvt_kernel<<<(int)(NW / 256), 256, 0, stream>>>(Wk, WkT, EMB, EMB);
  transpose_cvt_kernel<<<(int)(NW / 256), 256, 0, stream>>>(Wv, WvT, EMB, EMB);
  transpose_cvt_kernel<<<(int)(NW / 256), 256, 0, stream>>>(Wo, WoT, EMB, EMB);

  dim3 gg(EMB / 128, M / 128);                    // (4, 16)
  gemm_bb<1><<<gg, 256, 0, stream>>>(rowe, WqT, qw, M, EMB, EMB);
  gemm_bb<1><<<gg, 256, 0, stream>>>(cole, WkT, kwb, M, EMB, EMB);
  gemm_bb<2><<<gg, 256, 0, stream>>>(cole, WvT, vtw, M, EMB, EMB);

  attn_kernel<<<dim3(R_TOT / 16, B_TOT), 256, ATTN_SMEM_BYTES, stream>>>(
      qw, kwb, vtw, cost, mask, W1, W2, alpha, attb);

  gemm_bb<0><<<gg, 256, 0, stream>>>(attb, WoT, (float*)d_out, M, EMB, EMB);
}
